// InfiniteContextReader_69011534512414
// MI455X (gfx1250) — matrix operand order repaired and checked by reading, not compiled, not run on hardware
//
#include <hip/hip_runtime.h>
#include <cfloat>
#include <cmath>
#include <cstdint>

// ---------------------------------------------------------------------------
// Problem constants (from reference setup_inputs)
// ---------------------------------------------------------------------------
#define B_  8
#define S_  2048
#define H_  1024
#define D_  512
#define K_  8          // top_k
#define EPS 1e-8f

// GEMM tiling: 256 threads = 8 wave32; block tile 128x64; wave tile 32x32.
#define BLK_M 128
#define BLK_N 64
#define BLK_K 32
#define LDSW  56       // halves per LDS row: 112B = 16B-aligned, conflict-scattered

typedef __attribute__((ext_vector_type(16))) _Float16 v16h;
typedef __attribute__((ext_vector_type(8)))  _Float16 v8h;
typedef __attribute__((ext_vector_type(4)))  _Float16 v4h;
typedef __attribute__((ext_vector_type(8)))  float    v8f;

// ---------------------------------------------------------------------------
// CDNA5 async copy: global -> LDS, 16B per lane, tracked by ASYNCcnt.
// LDS aperture: low 32 bits of a generic LDS pointer == LDS byte offset.
// ---------------------------------------------------------------------------
__device__ __forceinline__ void async_copy_b128(void* lds_ptr, const void* gptr) {
    asm volatile("global_load_async_to_lds_b128 %0, %1, off"
                 :
                 : "v"((uint32_t)(uintptr_t)lds_ptr),
                   "v"((unsigned long long)(uintptr_t)gptr)
                 : "memory");
}
#define WAIT_ASYNCCNT(n) asm volatile("s_wait_asynccnt %0" ::"n"(n) : "memory")

// ---------------------------------------------------------------------------
// fp32 -> f16 elementwise convert (x -> x_h), float4 granularity
// ---------------------------------------------------------------------------
__global__ void cvt_f32_f16(const float* __restrict__ src,
                            _Float16* __restrict__ dst, int n4) {
    int i = blockIdx.x * blockDim.x + threadIdx.x;
    if (i >= n4) return;
    const float4 v = *(const float4*)(src + (size_t)i * 4);
    v4h h;
    h[0] = (_Float16)v.x; h[1] = (_Float16)v.y;
    h[2] = (_Float16)v.z; h[3] = (_Float16)v.w;
    *(v4h*)(dst + (size_t)i * 4) = h;
}

// ---------------------------------------------------------------------------
// Weight transpose + fp32 -> f16 convert:  dst[c*R + r] = (f16) src[r*C + c]
// ---------------------------------------------------------------------------
__global__ void transpose_cvt_f16(const float* __restrict__ src,
                                  _Float16* __restrict__ dst, int R, int C) {
    int id = blockIdx.x * blockDim.x + threadIdx.x;
    if (id >= R * C) return;
    int r = id % R;
    int c = id / R;
    dst[id] = (_Float16)src[(size_t)r * C + c];
}

// ---------------------------------------------------------------------------
// WMMA GEMM:  Out[M x N] = Ah[M x K](f16) * Bt[N x K](f16) (+ Res)
// Double-buffered LDS stages filled by async copies (ASYNCcnt), consumed by
// ds_load_b128 fragment reads feeding v_wmma_f32_16x16x32_f16.
// ---------------------------------------------------------------------------
template <bool RESIDUAL>
__global__ __launch_bounds__(256)
void wmma_gemm_f16(const _Float16* __restrict__ Ah,
                   const _Float16* __restrict__ Bt,
                   const float* __restrict__ Res,
                   float* __restrict__ Out,
                   int M, int Nn, int Kk) {
    __shared__ alignas(16) _Float16 As[2][BLK_M][LDSW];
    __shared__ alignas(16) _Float16 Bs[2][BLK_N][LDSW];

    const int tid  = threadIdx.x;
    const int lane = tid & 31;
    const int wave = tid >> 5;
    const int wm   = wave & 3;   // wave row 0..3 (32 rows each)
    const int wn   = wave >> 2;  // wave col 0..1 (32 cols each)
    const int l15  = lane & 15;
    const int hi   = lane >> 4;  // 0 | 1

    const int blockN = blockIdx.x * BLK_N;
    const int blockM = blockIdx.y * BLK_M;

    // Per-thread copy slots: A rows r0 and r0+64, B row r0; 16B chunk each.
    const int r0 = tid >> 2;          // 0..63
    const int co = (tid & 3) * 8;     // 0,8,16,24 (halves)

    v8f acc[2][2] = {};

    const int nk = Kk / BLK_K;

    // ---- stage k=0 into buffer 0 (3 async instructions per wave) ----
    {
        const int k0 = 0;
        async_copy_b128(&As[0][r0][co],      Ah + (size_t)(blockM + r0) * Kk + k0 + co);
        async_copy_b128(&As[0][r0 + 64][co], Ah + (size_t)(blockM + r0 + 64) * Kk + k0 + co);
        async_copy_b128(&Bs[0][r0][co],      Bt + (size_t)(blockN + r0) * Kk + k0 + co);
    }

    for (int k = 0; k < nk; ++k) {
        const int cur = k & 1;
        const int nxt = cur ^ 1;
        if (k + 1 < nk) {  // prefetch next stage while current computes
            const int k0 = (k + 1) * BLK_K;
            async_copy_b128(&As[nxt][r0][co],      Ah + (size_t)(blockM + r0) * Kk + k0 + co);
            async_copy_b128(&As[nxt][r0 + 64][co], Ah + (size_t)(blockM + r0 + 64) * Kk + k0 + co);
            async_copy_b128(&Bs[nxt][r0][co],      Bt + (size_t)(blockN + r0) * Kk + k0 + co);
            WAIT_ASYNCCNT(3);   // stage k done (async loads complete in order)
        } else {
            WAIT_ASYNCCNT(0);
        }
        __syncthreads();        // all waves' copies for stage k visible

        // --- fragments (ISA 7.12.2 16-bit layouts) ---
        v16h afrag[2], bfrag[2];
        {
            const int klo = hi * 8;           // A: lanes>=16 own K 8..15 / 24..31
#pragma unroll
            for (int im = 0; im < 2; ++im) {
                const int row = wm * 32 + im * 16 + l15;
                v8h lo = *(const v8h*)(&As[cur][row][klo]);
                v8h hh = *(const v8h*)(&As[cur][row][klo + 16]);
                afrag[im] = __builtin_shufflevector(lo, hh, 0, 1, 2, 3, 4, 5, 6, 7,
                                                    8, 9, 10, 11, 12, 13, 14, 15);
            }
            const int kb = hi * 8;            // B: lanes>=16 own K 8..15 / 24..31
#pragma unroll
            for (int in = 0; in < 2; ++in) {
                const int n = wn * 32 + in * 16 + l15;
                v8h lo = *(const v8h*)(&Bs[cur][n][kb]);
                v8h hh = *(const v8h*)(&Bs[cur][n][kb + 16]);
                bfrag[in] = __builtin_shufflevector(lo, hh, 0, 1, 2, 3, 4, 5, 6, 7,
                                                    8, 9, 10, 11, 12, 13, 14, 15);
            }
        }
#pragma unroll
        for (int im = 0; im < 2; ++im)
#pragma unroll
            for (int in = 0; in < 2; ++in)
                acc[im][in] = __builtin_amdgcn_wmma_f32_16x16x32_f16(
                    false, afrag[im], false, bfrag[in],
                    (short)0, acc[im][in], false, false);

        __syncthreads();        // readers done before buffer `cur` is refilled
    }

    // --- epilogue: C/D layout -> global (lane = N + 16*(M>=8), vgpr r = M%8) ---
#pragma unroll
    for (int im = 0; im < 2; ++im) {
#pragma unroll
        for (int in = 0; in < 2; ++in) {
            const int col = blockN + wn * 32 + in * 16 + l15;
#pragma unroll
            for (int r = 0; r < 8; ++r) {
                const int row = blockM + wm * 32 + im * 16 + r + hi * 8;
                const size_t o = (size_t)row * Nn + col;
                float v = acc[im][in][r];
                if (RESIDUAL) v += Res[o];
                Out[o] = v;
            }
        }
    }
}

// ---------------------------------------------------------------------------
// Pooled query: partial sums over S chunks (deterministic, no FP atomics)
// ---------------------------------------------------------------------------
__global__ void pool_partial(const float* __restrict__ qtok,
                             float* __restrict__ part) {
    const int b = blockIdx.y, c = blockIdx.x, t = threadIdx.x;
    for (int d = t; d < D_; d += 256) {
        float acc = 0.f;
        const float* p = qtok + ((size_t)b * S_ + (size_t)c * 128) * D_ + d;
        for (int r = 0; r < 128; ++r) acc += p[(size_t)r * D_];
        part[((size_t)b * 16 + c) * D_ + d] = acc;
    }
}

__global__ void pool_finish(const float* __restrict__ part,
                            float* __restrict__ qpool) {
    __shared__ float red[256];
    const int b = blockIdx.x, t = threadIdx.x;
    float v0 = 0.f, v1 = 0.f;
    for (int c = 0; c < 16; ++c) {
        v0 += part[((size_t)b * 16 + c) * D_ + t];
        v1 += part[((size_t)b * 16 + c) * D_ + t + 256];
    }
    v0 *= (1.0f / S_);
    v1 *= (1.0f / S_);
    red[t] = v0 * v0 + v1 * v1;
    __syncthreads();
    for (int s = 128; s > 0; s >>= 1) {
        if (t < s) red[t] += red[t + s];
        __syncthreads();
    }
    const float inv = 1.0f / (sqrtf(red[0]) + EPS);
    qpool[b * D_ + t]       = v0 * inv;
    qpool[b * D_ + t + 256] = v1 * inv;
}

// ---------------------------------------------------------------------------
// Scores: one wave per memory row; fused row-normalization.
// score[b][n] = dot(q[b], mem[n]) / (||mem[n]|| + eps)   (512 MB stream)
// ---------------------------------------------------------------------------
__global__ __launch_bounds__(256)
void scores_kernel(const float* __restrict__ mem, const float* __restrict__ qpool,
                   float* __restrict__ scores, int Nrows) {
    __shared__ float qs[B_][D_];
    const int t = threadIdx.x;
    for (int i = t; i < B_ * D_; i += 256) qs[i / D_][i % D_] = qpool[i];
    __syncthreads();

    const int lane = t & 31, wave = t >> 5;
    for (int n = blockIdx.x * 8 + wave; n < Nrows; n += gridDim.x * 8) {
        const float* row = mem + (size_t)n * D_;
        float nrm = 0.f;
        float dot[B_] = {};
#pragma unroll
        for (int j = 0; j < 4; ++j) {
            const int col = lane * 4 + j * 128;
            const float4 m4 = *(const float4*)(&row[col]);
            nrm += m4.x * m4.x + m4.y * m4.y + m4.z * m4.z + m4.w * m4.w;
#pragma unroll
            for (int b = 0; b < B_; ++b) {
                const float4 q4 = *(const float4*)(&qs[b][col]);
                dot[b] += m4.x * q4.x + m4.y * q4.y + m4.z * q4.z + m4.w * q4.w;
            }
        }
#pragma unroll
        for (int off = 16; off > 0; off >>= 1) {
            nrm += __shfl_xor(nrm, off, 32);
#pragma unroll
            for (int b = 0; b < B_; ++b) dot[b] += __shfl_xor(dot[b], off, 32);
        }
        if (lane == 0) {
            const float inv = 1.0f / (sqrtf(nrm) + EPS);
#pragma unroll
            for (int b = 0; b < B_; ++b)
                scores[(size_t)b * Nrows + n] = dot[b] * inv;
        }
    }
}

// ---------------------------------------------------------------------------
// Top-8 per batch: per-thread insertion top-8, then serial merge by thread 0.
// ---------------------------------------------------------------------------
__global__ void topk_kernel(const float* __restrict__ scores,
                            int* __restrict__ idxout, int Nrows) {
    __shared__ float sv[256 * K_];
    __shared__ int   si[256 * K_];
    const int b = blockIdx.x, t = threadIdx.x;
    float tv[K_];
    int   ti[K_];
#pragma unroll
    for (int k = 0; k < K_; ++k) { tv[k] = -FLT_MAX; ti[k] = 0x7fffffff; }
    for (int n = t; n < Nrows; n += 256) {
        const float v = scores[(size_t)b * Nrows + n];
        if (v > tv[K_ - 1] || (v == tv[K_ - 1] && n < ti[K_ - 1])) {
            int pos = K_ - 1;
            while (pos > 0 &&
                   (v > tv[pos - 1] || (v == tv[pos - 1] && n < ti[pos - 1]))) {
                tv[pos] = tv[pos - 1]; ti[pos] = ti[pos - 1]; --pos;
            }
            tv[pos] = v; ti[pos] = n;
        }
    }
#pragma unroll
    for (int k = 0; k < K_; ++k) { sv[t * K_ + k] = tv[k]; si[t * K_ + k] = ti[k]; }
    __syncthreads();
    if (t == 0) {
        float fv[K_];
        int   fi[K_];
#pragma unroll
        for (int k = 0; k < K_; ++k) { fv[k] = -FLT_MAX; fi[k] = 0x7fffffff; }
        for (int j = 0; j < 256 * K_; ++j) {
            const float v = sv[j];
            const int   n = si[j];
            if (v > fv[K_ - 1] || (v == fv[K_ - 1] && n < fi[K_ - 1])) {
                int pos = K_ - 1;
                while (pos > 0 &&
                       (v > fv[pos - 1] || (v == fv[pos - 1] && n < fi[pos - 1]))) {
                    fv[pos] = fv[pos - 1]; fi[pos] = fi[pos - 1]; --pos;
                }
                fv[pos] = v; fi[pos] = n;
            }
        }
#pragma unroll
        for (int k = 0; k < K_; ++k) idxout[b * K_ + k] = fi[k];
    }
}

// ---------------------------------------------------------------------------
// Gather + normalize retrieved rows: retr[b][k][:] = mem[idx] / (||mem[idx]||+eps)
// ---------------------------------------------------------------------------
__global__ void gather_norm(const float* __restrict__ mem,
                            const int* __restrict__ idx,
                            float* __restrict__ retr) {
    __shared__ float red[256];
    const int b = blockIdx.y, k = blockIdx.x, t = threadIdx.x;
    const int row = idx[b * K_ + k];
    const float* src = mem + (size_t)row * D_;
    const float m0 = src[t], m1 = src[t + 256];
    red[t] = m0 * m0 + m1 * m1;
    __syncthreads();
    for (int s = 128; s > 0; s >>= 1) {
        if (t < s) red[t] += red[t + s];
        __syncthreads();
    }
    const float inv = 1.0f / (sqrtf(red[0]) + EPS);
    float* dst = retr + ((size_t)b * K_ + k) * D_;
    dst[t]       = m0 * inv;
    dst[t + 256] = m1 * inv;
}

// ---------------------------------------------------------------------------
// Per-token 8-way attention; ctx emitted directly as f16 (GEMM2 A operand).
// ---------------------------------------------------------------------------
__global__ __launch_bounds__(256)
void attn_kernel(const float* __restrict__ qtok, const float* __restrict__ retr,
                 _Float16* __restrict__ ctx_h) {
    __shared__ float rs[K_][D_];
    const int b = blockIdx.y, t = threadIdx.x;
    for (int i = t; i < K_ * D_; i += 256)
        rs[i / D_][i % D_] = retr[(size_t)b * K_ * D_ + i];
    __syncthreads();

    const int lane = t & 31, wave = t >> 5;
    const int s = blockIdx.x * 8 + wave;
    const float* q = qtok + ((size_t)b * S_ + s) * D_;
    _Float16* cd = ctx_h + ((size_t)b * S_ + s) * D_;

    float dot[K_] = {};
#pragma unroll
    for (int j = 0; j < 4; ++j) {
        const int col = lane * 4 + j * 128;
        const float4 q4 = *(const float4*)(&q[col]);
#pragma unroll
        for (int k = 0; k < K_; ++k) {
            const float4 r4 = *(const float4*)(&rs[k][col]);
            dot[k] += q4.x * r4.x + q4.y * r4.y + q4.z * r4.z + q4.w * r4.w;
        }
    }
#pragma unroll
    for (int off = 16; off > 0; off >>= 1)
#pragma unroll
        for (int k = 0; k < K_; ++k) dot[k] += __shfl_xor(dot[k], off, 32);

    const float scale = 1.0f / sqrtf((float)D_);
    float l[K_], mx = -FLT_MAX;
#pragma unroll
    for (int k = 0; k < K_; ++k) { l[k] = dot[k] * scale; mx = fmaxf(mx, l[k]); }
    float se = 0.f;
#pragma unroll
    for (int k = 0; k < K_; ++k) { l[k] = expf(l[k] - mx); se += l[k]; }
    const float invs = 1.0f / se;

#pragma unroll
    for (int j = 0; j < 4; ++j) {
        const int col = lane * 4 + j * 128;
        float cx = 0.f, cy = 0.f, cz = 0.f, cw = 0.f;
#pragma unroll
        for (int k = 0; k < K_; ++k) {
            const float4 r4 = *(const float4*)(&rs[k][col]);
            const float w = l[k] * invs;
            cx += w * r4.x; cy += w * r4.y; cz += w * r4.z; cw += w * r4.w;
        }
        v4h h;
        h[0] = (_Float16)cx; h[1] = (_Float16)cy;
        h[2] = (_Float16)cz; h[3] = (_Float16)cw;
        *(v4h*)(&cd[col]) = h;
    }
}

// ---------------------------------------------------------------------------
// kernel_launch
// ---------------------------------------------------------------------------
extern "C" void kernel_launch(void* const* d_in, const int* in_sizes, int n_in,
                              void* d_out, int out_size, void* d_ws, size_t ws_size,
                              hipStream_t stream) {
    const float* x   = (const float*)d_in[0];   // [B,S,H]
    const float* mem = (const float*)d_in[1];   // [N,D]
    const float* Wq  = (const float*)d_in[2];   // [H,D]
    const float* Wo  = (const float*)d_in[3];   // [D,H]
    const int Nrows  = in_sizes[1] / D_;

    // workspace layout (all 256B-aligned sizes)
    char* ws = (char*)d_ws;
    float*    qtok  = (float*)ws;    ws += (size_t)B_ * S_ * D_ * 4;   // 32MB
    _Float16* xh    = (_Float16*)ws; ws += (size_t)B_ * S_ * H_ * 2;   // 32MB; ctx_h aliases (xh dead after GEMM1)
    _Float16* ctxh  = xh;                                              // 16MB alias
    _Float16* WqT   = (_Float16*)ws; ws += (size_t)D_ * H_ * 2;        // 1MB  [D][H] = NxK
    _Float16* WoT   = (_Float16*)ws; ws += (size_t)H_ * D_ * 2;        // 1MB  [H][D] = NxK
    float*    part  = (float*)ws;    ws += (size_t)B_ * 16 * D_ * 4;   // 256KB
    float*    qpool = (float*)ws;    ws += (size_t)B_ * D_ * 4;        // 16KB
    float*    scr   = (float*)ws;    ws += (size_t)B_ * Nrows * 4;     // 8MB
    int*      topk  = (int*)ws;      ws += 256;
    float*    retr  = (float*)ws;                                      // 128KB

    // 1) precision staging: x -> f16, weights transpose+convert to NxK f16
    cvt_f32_f16<<<(B_ * S_ * H_ / 4 + 255) / 256, 256, 0, stream>>>(
        x, xh, B_ * S_ * H_ / 4);
    transpose_cvt_f16<<<(H_ * D_ + 255) / 256, 256, 0, stream>>>(Wq, WqT, H_, D_);
    transpose_cvt_f16<<<(D_ * H_ + 255) / 256, 256, 0, stream>>>(Wo, WoT, D_, H_);

    // 2) q_tok = x @ Wq   (M=16384, N=512, K=1024): async-LDS + WMMA
    wmma_gemm_f16<false><<<dim3(D_ / BLK_N, (B_ * S_) / BLK_M), 256, 0, stream>>>(
        xh, WqT, nullptr, qtok, B_ * S_, D_, H_);

    // 3) pooled, normalized query
    pool_partial<<<dim3(16, B_), 256, 0, stream>>>(qtok, part);
    pool_finish<<<B_, 256, 0, stream>>>(part, qpool);

    // 4) cosine scores over the 512MB archive (bandwidth-bound VALU pass)
    scores_kernel<<<2048, 256, 0, stream>>>(mem, qpool, scr, Nrows);

    // 5) top-8 + gather/normalize retrieved rows
    topk_kernel<<<B_, 256, 0, stream>>>(scr, topk, Nrows);
    gather_norm<<<dim3(K_, B_), 256, 0, stream>>>(mem, topk, retr);

    // 6) per-token attention -> ctx in f16 (aliases x_h, which is now dead)
    attn_kernel<<<dim3(S_ / 8, B_), 256, 0, stream>>>(qtok, retr, ctxh);

    // 7) out = x + ctx @ Wo  (M=16384, N=1024, K=512): async-LDS + WMMA + residual
    wmma_gemm_f16<true><<<dim3(H_ / BLK_N, (B_ * S_) / BLK_M), 256, 0, stream>>>(
        ctxh, WoT, x, (float*)d_out, B_ * S_, H_, D_);
}
